// KGramMLPSeqModel_17274358464625
// MI455X (gfx1250) — matrix operand order repaired and checked by reading, not compiled, not run on hardware
//
#include <hip/hip_runtime.h>

// ---- problem constants (from reference) ----
#define VOCAB 32000
#define KCTX  3
#define EMBED 1024
#define SEQ   2048
#define BATCH 2
#define MROWS (SEQ * BATCH)   // 4096

// ---- GEMM tiling ----
#define BM  128
#define BN  256
#define BK  32
#define LDA 48                // padded LDS row stride (elements): 96 B, 16B aligned

typedef __attribute__((ext_vector_type(16))) __bf16         v16bf;
typedef __attribute__((ext_vector_type(8)))  float          v8f;
typedef __attribute__((ext_vector_type(4)))  int            v4i;
typedef __attribute__((ext_vector_type(4)))  unsigned int   u32x4;
typedef __attribute__((ext_vector_type(4)))  float          f32x4;
typedef __attribute__((ext_vector_type(4)))  unsigned short u16x4;
typedef __attribute__((ext_vector_type(8)))  unsigned short u16x8;

#define AS3 __attribute__((address_space(3)))
#define AS1 __attribute__((address_space(1)))

// CDNA5 async global->LDS copy path (guarded: compiles cleanly either way)
#if defined(__HIP_DEVICE_COMPILE__)
#if __has_builtin(__builtin_amdgcn_global_load_async_to_lds_b128) && \
    __has_builtin(__builtin_amdgcn_s_wait_asynccnt)
#define HAVE_ASYNC 1
#else
#define HAVE_ASYNC 0
#endif
#else
#define HAVE_ASYNC 0
#endif

union FragCast { u32x4 u[2]; v16bf v; };

static __device__ __forceinline__ unsigned short f2bf(float f) {
    // round-to-nearest-even f32 -> bf16
    unsigned u = __builtin_bit_cast(unsigned, f);
    u += 0x7fffu + ((u >> 16) & 1u);
    return (unsigned short)(u >> 16);
}

// --------------------------------------------------------------------------
// Kernel 1: h[m, e] = silu(b1[e] + sum_i W1[e, i*VOCAB + ctx_i(m)]) -> bf16
// --------------------------------------------------------------------------
__global__ void __launch_bounds__(256)
kgram_build_h(const int* __restrict__ tok,
              const float* __restrict__ W1,
              const float* __restrict__ b1,
              unsigned short* __restrict__ h) {
    const int m  = blockIdx.x;          // 0..4095  (m = s*BATCH + b)
    const int e0 = threadIdx.x * 4;     // 0..1020
    const int s  = m >> 1;

    int col[KCTX];
#pragma unroll
    for (int i = 0; i < KCTX; ++i) {
        const int d = KCTX - i;                       // lookback distance
        const int t = (s >= d) ? tok[m - d * BATCH] : 0;
        col[i] = i * VOCAB + t;
    }

    f32x4 acc = *(const f32x4*)(b1 + e0);
    u16x4 outv;
#pragma unroll
    for (int j = 0; j < 4; ++j) {
        const int base = (e0 + j) * (KCTX * VOCAB);   // row of W1 (EMBED x 96000)
        float x = acc[j] + W1[base + col[0]] + W1[base + col[1]] + W1[base + col[2]];
        x = x / (1.0f + __expf(-x));                  // silu
        outv[j] = f2bf(x);
    }
    *(u16x4*)(h + m * EMBED + e0) = outv;
}

// --------------------------------------------------------------------------
// Kernel 1b: one-time W2 f32 -> bf16 (only when workspace is large enough)
// --------------------------------------------------------------------------
__global__ void __launch_bounds__(256)
kgram_convert_w2(const float* __restrict__ W2, unsigned short* __restrict__ W2b) {
    const size_t i = ((size_t)blockIdx.x * 256 + threadIdx.x) * 8;
    f32x4 a = *(const f32x4*)(W2 + i);
    f32x4 b = *(const f32x4*)(W2 + i + 4);
    u16x8 o;
    o[0] = f2bf(a[0]); o[1] = f2bf(a[1]); o[2] = f2bf(a[2]); o[3] = f2bf(a[3]);
    o[4] = f2bf(b[0]); o[5] = f2bf(b[1]); o[6] = f2bf(b[2]); o[7] = f2bf(b[3]);
    *(u16x8*)(W2b + i) = o;
}

// --------------------------------------------------------------------------
// Kernel 2: out[m, v] = sum_e h[m,e] * W2[v,e] + b2[v]
// bf16 WMMA (16x16x32), f32 accumulation. Block tile 128x256, K-step 32.
// PREB: B matrix already bf16 in workspace -> pure-copy staging (async-capable)
// --------------------------------------------------------------------------
template <bool PREB>
__global__ void __launch_bounds__(256)
kgram_logits_gemm(const unsigned short* __restrict__ h,   // bf16 bits, MROWS x EMBED
                  const float* __restrict__ W2f,          // VOCAB x EMBED f32
                  const unsigned short* __restrict__ W2b, // VOCAB x EMBED bf16 (PREB)
                  const float* __restrict__ b2,
                  float* __restrict__ out) {
    __shared__ __attribute__((aligned(16))) unsigned short Alds[BM * LDA]; // 12 KB
    __shared__ __attribute__((aligned(16))) unsigned short Blds[BN * LDA]; // 24 KB

    const int tid  = threadIdx.x;
    const int lane = tid & 31;
    const int wave = tid >> 5;          // 0..7
    const int r    = lane & 15;
    const int hi   = lane >> 4;         // half-wave select
    const int mBase = (wave & 1) * 64;  // wave grid 2 (M) x 4 (N)
    const int nBase = (wave >> 1) * 64;
    const int mBlock = blockIdx.y * BM;
    const int nBlock = blockIdx.x * BN;

    v8f acc[4][4] = {};

    for (int k0 = 0; k0 < EMBED; k0 += BK) {
        // ---- stage A tile (bf16 h): 128 rows x 32, 16B chunks ----
        {
            int c = tid;                               // 512 chunks total
#pragma unroll
            for (int p = 0; p < 2; ++p, c += 256) {
                const int row = c >> 2, seg = c & 3;
                const unsigned short* gsrc = h + (size_t)(mBlock + row) * EMBED + k0 + seg * 8;
                unsigned short* ldst = &Alds[row * LDA + seg * 8];
#if HAVE_ASYNC
                __builtin_amdgcn_global_load_async_to_lds_b128(
                    (AS1 v4i*)gsrc, (AS3 v4i*)ldst, 0, 0);
#else
                *(u32x4*)ldst = *(const u32x4*)gsrc;
#endif
            }
        }
        // ---- stage B tile: 256 rows x 32 ----
        if constexpr (PREB) {
            int c = tid;                               // 1024 chunks of 16B
#pragma unroll
            for (int p = 0; p < 4; ++p, c += 256) {
                const int row = c >> 2, seg = c & 3;
                const unsigned short* gsrc = W2b + (size_t)(nBlock + row) * EMBED + k0 + seg * 8;
                unsigned short* ldst = &Blds[row * LDA + seg * 8];
#if HAVE_ASYNC
                __builtin_amdgcn_global_load_async_to_lds_b128(
                    (AS1 v4i*)gsrc, (AS3 v4i*)ldst, 0, 0);
#else
                *(u32x4*)ldst = *(const u32x4*)gsrc;
#endif
            }
        } else {
            const float* src = W2f + (size_t)(nBlock + tid) * EMBED + k0;
#pragma unroll
            for (int q = 0; q < 8; ++q) {
                f32x4 v = *(const f32x4*)(src + q * 4);
                u16x4 b;
                b[0] = f2bf(v[0]); b[1] = f2bf(v[1]);
                b[2] = f2bf(v[2]); b[3] = f2bf(v[3]);
                *(u16x4*)(&Blds[tid * LDA + q * 4]) = b;
            }
            if (k0 + BK < EMBED) __builtin_prefetch(src + BK, 0, 0); // next K slice
        }
#if HAVE_ASYNC
        __builtin_amdgcn_s_wait_asynccnt(0);
#endif
        __syncthreads();

        // ---- fragment loads (ISA 16-bit A/B wave32 layouts) ----
        FragCast af[4], bfg[4];
#pragma unroll
        for (int mt = 0; mt < 4; ++mt) {
            const int base = (mBase + mt * 16 + r) * LDA + hi * 8;
            af[mt].u[0] = *(const u32x4*)(&Alds[base]);        // K = hi*8 + 0..7
            af[mt].u[1] = *(const u32x4*)(&Alds[base + 16]);   // K = 16 + hi*8 + 0..7
        }
#pragma unroll
        for (int nt = 0; nt < 4; ++nt) {
            const int base = (nBase + nt * 16 + r) * LDA + hi * 8;
            bfg[nt].u[0] = *(const u32x4*)(&Blds[base]);       // K = hi*8 + 0..7
            bfg[nt].u[1] = *(const u32x4*)(&Blds[base + 16]);  // K = 16 + hi*8 + 0..7
        }

        // ---- 16 WMMAs per wave per K-step ----
#pragma unroll
        for (int mt = 0; mt < 4; ++mt)
#pragma unroll
            for (int nt = 0; nt < 4; ++nt)
                acc[mt][nt] = __builtin_amdgcn_wmma_f32_16x16x32_bf16(
                    false, af[mt].v, false, bfg[nt].v,
                    (short)0, acc[mt][nt], false, false);

        __syncthreads();
    }

    // ---- epilogue: + b2, store f32 (C/D layout: VGPR i -> row i + 8*hi) ----
#pragma unroll
    for (int nt = 0; nt < 4; ++nt) {
        const int gn = nBlock + nBase + nt * 16 + r;
        const float bias = b2[gn];
#pragma unroll
        for (int mt = 0; mt < 4; ++mt) {
            const int gm0 = mBlock + mBase + mt * 16 + hi * 8;
#pragma unroll
            for (int i = 0; i < 8; ++i)
                out[(size_t)(gm0 + i) * VOCAB + gn] = acc[mt][nt][i] + bias;
        }
    }
}

// --------------------------------------------------------------------------
extern "C" void kernel_launch(void* const* d_in, const int* in_sizes, int n_in,
                              void* d_out, int out_size, void* d_ws, size_t ws_size,
                              hipStream_t stream) {
    const int*   tok = (const int*)d_in[0];     // (SEQ, BATCH) int32
    const float* W1  = (const float*)d_in[1];   // (EMBED, K*VOCAB)
    const float* b1  = (const float*)d_in[2];   // (EMBED,)
    const float* W2  = (const float*)d_in[3];   // (VOCAB, EMBED)
    const float* b2  = (const float*)d_in[4];   // (VOCAB,)
    float* out = (float*)d_out;                 // (SEQ, BATCH, VOCAB) f32

    const size_t hElems  = (size_t)MROWS * EMBED;         // 4.19M bf16 = 8 MB
    const size_t w2Elems = (size_t)VOCAB * EMBED;         // 32.77M bf16 = 65.5 MB
    unsigned short* hbuf = (unsigned short*)d_ws;

    kgram_build_h<<<MROWS, 256, 0, stream>>>(tok, W1, b1, hbuf);

    dim3 grid(VOCAB / BN, MROWS / BM);   // (125, 32)
    if (ws_size >= (hElems + w2Elems) * sizeof(unsigned short)) {
        // big workspace: pre-convert W2 once, GEMM stages pure bf16 copies
        unsigned short* w2b = hbuf + hElems;
        kgram_convert_w2<<<(int)(w2Elems / (256 * 8)), 256, 0, stream>>>(W2, w2b);
        kgram_logits_gemm<true><<<grid, 256, 0, stream>>>(hbuf, W2, w2b, b2, out);
    } else {
        // small workspace: convert W2 tiles on the fly (register staging)
        kgram_logits_gemm<false><<<grid, 256, 0, stream>>>(hbuf, W2, nullptr, b2, out);
    }
}
